// CausalAttention_56160992362649
// MI455X (gfx1250) — compile-verified, matrix operand order checked by reading, not run on hardware
//
#include <hip/hip_runtime.h>

// ---------------------------------------------------------------------------
// CausalAttention for MI455X (gfx1250): bf16 WMMA + async-to-LDS staging.
//   x: (4,2048,2048) f32   W_qkv: (6144,2048) f32   W_out: (2048,2048) f32
//   out: (4,2048,2048) f32
// ---------------------------------------------------------------------------

#define B_SZ   4
#define T_SZ   2048
#define D_SZ   2048
#define H_SZ   16
#define DH_SZ  128
#define E_SZ   (3 * D_SZ)      // 6144
#define M_SZ   (B_SZ * T_SZ)   // 8192

typedef __attribute__((ext_vector_type(16))) __bf16 v16bf;
typedef __attribute__((ext_vector_type(8)))  float  v8f;
typedef __attribute__((ext_vector_type(4)))  unsigned int v4u;

union Frag16 { v4u u[2]; v16bf v; };

__device__ __forceinline__ unsigned short f2bf(float f) {
  unsigned int u = __float_as_uint(f);
  u = (u + 0x7FFFu + ((u >> 16) & 1u)) >> 16;   // round-to-nearest-even
  return (unsigned short)u;
}

// bf16 WMMA fragment load (A layout; mirrored B layout has identical per-lane
// addressing with row := column): two contiguous 16B loads per lane.
__device__ __forceinline__ v16bf ld_frag_row(const unsigned short* __restrict__ rp,
                                             int k0, int kg) {
  const unsigned short* p = rp + k0 + (kg << 3);
  Frag16 f;
  f.u[0] = *(const v4u*)(p);
  f.u[1] = *(const v4u*)(p + 16);
  return f.v;
}
__device__ __forceinline__ v16bf load_frag(const unsigned short* __restrict__ p,
                                           int ld, int row, int k0, int kg) {
  return ld_frag_row(p + (size_t)row * ld, k0, kg);
}

__device__ __forceinline__ v8f wmma_bf16(v16bf a, v16bf b, v8f c) {
  return __builtin_amdgcn_wmma_f32_16x16x32_bf16(false, a, false, b,
                                                 (short)0, c, false, false);
}

// ---- CDNA5 async global->LDS copy (ASYNCcnt-tracked), per ISA ch.10/15.18 --
__device__ __forceinline__ void async_cp_b128(unsigned lds_addr,
                                              unsigned long long gaddr) {
  asm volatile("global_load_async_to_lds_b128 %0, %1, off"
               :: "v"(lds_addr), "v"(gaddr) : "memory");
}
#define WAIT_ASYNCCNT(n) asm volatile("s_wait_asynccnt " #n ::: "memory")

// ---------------------------------------------------------------------------
// Kernel 1: fp32 -> bf16 conversion (grid-stride)
// ---------------------------------------------------------------------------
__global__ void ca_cvt_bf16(const float* __restrict__ in,
                            unsigned short* __restrict__ out, int n) {
  int i = blockIdx.x * blockDim.x + threadIdx.x;
  int stride = gridDim.x * blockDim.x;
  for (; i < n; i += stride) out[i] = f2bf(in[i]);
}

// ---------------------------------------------------------------------------
// Kernel 2: QKV projection. C[m,e] = sum_d A[m,d] * W[e,d]
// Wave tile 32x64, software-pipelined k loop (next fragments load while the
// current 8 WMMAs issue). Scatter into Q/K (b,h,t,dh) and V^T (b,h,dh,t).
// block = 128 (4 waves along n). grid = (E/256, M/32).
// ---------------------------------------------------------------------------
__global__ void ca_gemm_qkv(const unsigned short* __restrict__ A,
                            const unsigned short* __restrict__ W,
                            unsigned short* __restrict__ Qb,
                            unsigned short* __restrict__ Kb,
                            unsigned short* __restrict__ Vtb) {
  const int lane = threadIdx.x & 31;
  const int wave = threadIdx.x >> 5;
  const int kg = lane >> 4, ln = lane & 15;
  const int m0 = blockIdx.y * 32;
  const int n0 = (blockIdx.x * 4 + wave) * 64;

  const unsigned short* A0 = A + (size_t)(m0 + ln) * D_SZ;
  const unsigned short* A1 = A0 + (size_t)16 * D_SZ;
  const unsigned short* W0 = W + (size_t)(n0 + ln) * D_SZ;

  v8f acc[2][4] = {{v8f{}, v8f{}, v8f{}, v8f{}}, {v8f{}, v8f{}, v8f{}, v8f{}}};

  v16bf a0 = ld_frag_row(A0, 0, kg);
  v16bf a1 = ld_frag_row(A1, 0, kg);
  v16bf b0 = ld_frag_row(W0 + (size_t)0 * 16 * D_SZ, 0, kg);
  v16bf b1 = ld_frag_row(W0 + (size_t)1 * 16 * D_SZ, 0, kg);
  v16bf b2 = ld_frag_row(W0 + (size_t)2 * 16 * D_SZ, 0, kg);
  v16bf b3 = ld_frag_row(W0 + (size_t)3 * 16 * D_SZ, 0, kg);

  for (int k0 = 0; k0 < D_SZ; k0 += 32) {
    const int kn = (k0 + 32 < D_SZ) ? k0 + 32 : 0;   // branch-free pipeline
    v16bf na0 = ld_frag_row(A0, kn, kg);
    v16bf na1 = ld_frag_row(A1, kn, kg);
    v16bf nb0 = ld_frag_row(W0 + (size_t)0 * 16 * D_SZ, kn, kg);
    v16bf nb1 = ld_frag_row(W0 + (size_t)1 * 16 * D_SZ, kn, kg);
    v16bf nb2 = ld_frag_row(W0 + (size_t)2 * 16 * D_SZ, kn, kg);
    v16bf nb3 = ld_frag_row(W0 + (size_t)3 * 16 * D_SZ, kn, kg);

    acc[0][0] = wmma_bf16(a0, b0, acc[0][0]);
    acc[1][0] = wmma_bf16(a1, b0, acc[1][0]);
    acc[0][1] = wmma_bf16(a0, b1, acc[0][1]);
    acc[1][1] = wmma_bf16(a1, b1, acc[1][1]);
    acc[0][2] = wmma_bf16(a0, b2, acc[0][2]);
    acc[1][2] = wmma_bf16(a1, b2, acc[1][2]);
    acc[0][3] = wmma_bf16(a0, b3, acc[0][3]);
    acc[1][3] = wmma_bf16(a1, b3, acc[1][3]);

    a0 = na0; a1 = na1; b0 = nb0; b1 = nb1; b2 = nb2; b3 = nb3;
  }

#pragma unroll
  for (int t = 0; t < 4; ++t) {
    const int e = n0 + t * 16 + ln;
    const int s  = e >> 11;          // 0=Q 1=K 2=V
    const int h  = (e >> 7) & (H_SZ - 1);
    const int dh = e & (DH_SZ - 1);
#pragma unroll
    for (int mi = 0; mi < 2; ++mi)
#pragma unroll
      for (int r = 0; r < 8; ++r) {
        const int m  = m0 + mi * 16 + r + kg * 8;
        const int bi = m >> 11;
        const int tt = m & (T_SZ - 1);
        const size_t bhh = (size_t)(bi * H_SZ + h);
        const unsigned short val = f2bf(acc[mi][t][r]);
        if (s == 0)      Qb [(bhh * T_SZ + tt) * DH_SZ + dh] = val;
        else if (s == 1) Kb [(bhh * T_SZ + tt) * DH_SZ + dh] = val;
        else             Vtb[(bhh * DH_SZ + dh) * T_SZ + tt] = val;
      }
  }
}

// ---------------------------------------------------------------------------
// Kernel 3: causal flash attention.
// Block = 128 threads (4 waves) = 4 consecutive 16-row q tiles of one (b,h).
// Per 32-key chunk the block async-copies the K tile (32x128, contiguous) and
// V^T tile (128x32) into double-buffered LDS; each wave runs S = Q K^T,
// online softmax, and O += P V against the shared staged tiles.
// grid = (T/64, H, B).
// ---------------------------------------------------------------------------
__global__ __launch_bounds__(128) void ca_attn(
    const unsigned short* __restrict__ Qb,
    const unsigned short* __restrict__ Kb,
    const unsigned short* __restrict__ Vtb,
    unsigned short* __restrict__ Ob) {
  __shared__ __align__(16) unsigned short KbufL[2][32 * DH_SZ];   // 2 x 8KB
  __shared__ __align__(16) unsigned short VbufL[2][DH_SZ * 32];   // 2 x 8KB
  __shared__ __align__(16) unsigned short pTile[4][16 * 32];      // per-wave

  const int tid  = threadIdx.x;
  const int lane = tid & 31, wave = tid >> 5;
  const int kg = lane >> 4, ln = lane & 15;
  const int h = blockIdx.y, bi = blockIdx.z;
  const int qt = blockIdx.x * 4 + wave;                 // this wave's q tile
  const size_t bh = (size_t)(bi * H_SZ + h);

  const unsigned short* Qp = Qb  + (bh * T_SZ + (size_t)qt * 16) * DH_SZ;
  const unsigned short* Kp = Kb  +  bh * T_SZ * DH_SZ;
  const unsigned short* Vp = Vtb +  bh * DH_SZ * T_SZ;

  // causal extent of the whole block (last tile): exactly 2*(bx+1) chunks
  const int nchunks = 2 * (blockIdx.x + 1);

  const unsigned klds[2] = {(unsigned)(unsigned long long)&KbufL[0][0],
                            (unsigned)(unsigned long long)&KbufL[1][0]};
  const unsigned vlds[2] = {(unsigned)(unsigned long long)&VbufL[0][0],
                            (unsigned)(unsigned long long)&VbufL[1][0]};

  // Stage one 32-key chunk: 128 threads x 4 x b128 each for K and for V.
  auto stage = [&](int j0, int buf) {
    unsigned lk = klds[buf] + tid * 16;
    unsigned long long gk =
        (unsigned long long)(Kp + (size_t)j0 * DH_SZ + tid * 8);   // contiguous
    unsigned lv = vlds[buf] + tid * 16;
    unsigned long long gv =
        (unsigned long long)(Vp + (size_t)(tid >> 2) * T_SZ + (tid & 3) * 8 + j0);
#pragma unroll
    for (int i = 0; i < 4; ++i) {
      async_cp_b128(lk, gk); lk += 2048; gk += 2048;               // +128 chunks
      async_cp_b128(lv, gv); lv += 2048; gv += (unsigned long long)32 * T_SZ * 2;
    }
  };

  stage(0, 0);   // prologue: chunk 0 -> buffer 0 (8 async ops per wave)

  // Q fragments: 16x128 = 4 k-steps of 32 (regular loads, overlap with async)
  v16bf qf[4];
#pragma unroll
  for (int i = 0; i < 4; ++i) qf[i] = load_frag(Qp, DH_SZ, ln, i * 32, kg);

  v8f o[8] = {v8f{}, v8f{}, v8f{}, v8f{}, v8f{}, v8f{}, v8f{}, v8f{}};
  float mrow[8], lrow[8];
#pragma unroll
  for (int r = 0; r < 8; ++r) { mrow[r] = -3.0e38f; lrow[r] = 0.0f; }

  const float sl2 = 0.08838834764831845f * 1.4426950408889634f;  // scale*log2e

  for (int c = 0; c < nchunks; ++c) {
    const int j0 = c * 32;
    const int buf = c & 1;

    if (c + 1 < nchunks) {
      stage(j0 + 32, buf ^ 1);      // prefetch next chunk into other buffer
      WAIT_ASYNCCNT(0x8);           // this wave's chunk-c copies complete
    } else {
      WAIT_ASYNCCNT(0x0);
    }
    __syncthreads();                // all waves' chunk-c copies complete

    const unsigned short* Kl = &KbufL[buf][0];
    const unsigned short* Vl = &VbufL[buf][0];

    // ---- scores: two 16x16 subtiles over 32 keys, from LDS ----
    v8f s0 = {}, s1 = {};
#pragma unroll
    for (int i = 0; i < 4; ++i) {
      v16bf kb0 = load_frag(Kl, DH_SZ, ln,      i * 32, kg);
      v16bf kb1 = load_frag(Kl, DH_SZ, 16 + ln, i * 32, kg);
      s0 = wmma_bf16(qf[i], kb0, s0);
      s1 = wmma_bf16(qf[i], kb1, s1);
    }

    // ---- online softmax ----
    float ex0[8], ex1[8], alpha[8];
#pragma unroll
    for (int r = 0; r < 8; ++r) {
      const int qrow = qt * 16 + r + kg * 8;
      const float v0 = (j0 + ln      <= qrow) ? s0[r] * sl2 : -3.0e38f;
      const float v1 = (j0 + 16 + ln <= qrow) ? s1[r] * sl2 : -3.0e38f;
      float mx = fmaxf(v0, v1);
#pragma unroll
      for (int d = 8; d >= 1; d >>= 1) mx = fmaxf(mx, __shfl_xor(mx, d, 32));
      const float mnew = fmaxf(mrow[r], mx);
      alpha[r] = exp2f(mrow[r] - mnew);
      mrow[r] = mnew;
      const float e0 = exp2f(v0 - mnew);
      const float e1 = exp2f(v1 - mnew);
      float rs = e0 + e1;
#pragma unroll
      for (int d = 8; d >= 1; d >>= 1) rs += __shfl_xor(rs, d, 32);
      lrow[r] = lrow[r] * alpha[r] + rs;
      ex0[r] = e0; ex1[r] = e1;
    }
#pragma unroll
    for (int t = 0; t < 8; ++t)
#pragma unroll
      for (int r = 0; r < 8; ++r) o[t][r] *= alpha[r];

    // ---- transpose P (C layout -> A layout) through per-wave LDS ----
#pragma unroll
    for (int r = 0; r < 8; ++r) {
      const int rr = r + kg * 8;
      pTile[wave][rr * 32 + ln]      = f2bf(ex0[r]);
      pTile[wave][rr * 32 + 16 + ln] = f2bf(ex1[r]);
    }
    __syncthreads();
    Frag16 pf;
    {
      const unsigned short* pr = &pTile[wave][ln * 32 + kg * 8];
      pf.u[0] = *(const v4u*)(pr);
      pf.u[1] = *(const v4u*)(pr + 16);
    }

    // ---- O(16x128) += P(16x32) * V(32x128), V^T tile from LDS ----
#pragma unroll
    for (int t = 0; t < 8; ++t) {
      v16bf vbf = load_frag(Vl, 32, t * 16 + ln, 0, kg);
      o[t] = wmma_bf16(pf.v, vbf, o[t]);
    }
    __syncthreads();   // all readers done before next chunk overwrites buffer
  }

  // ---- normalize and store O as bf16 in (b, t, h*128+dh) layout ----
  float inv[8];
#pragma unroll
  for (int r = 0; r < 8; ++r) inv[r] = 1.0f / lrow[r];
#pragma unroll
  for (int t = 0; t < 8; ++t)
#pragma unroll
    for (int r = 0; r < 8; ++r) {
      const int trow = qt * 16 + r + kg * 8;
      const size_t idx =
          ((size_t)(bi * T_SZ + trow) * D_SZ) + h * DH_SZ + t * 16 + ln;
      Ob[idx] = f2bf(o[t][r] * inv[r]);
    }
}

// ---------------------------------------------------------------------------
// Kernel 4: output projection, wave tile 32x64, pipelined, f32 store.
// grid = (D/256, M/32), block = 128.
// ---------------------------------------------------------------------------
__global__ void ca_gemm_out(const unsigned short* __restrict__ A,
                            const unsigned short* __restrict__ W,
                            float* __restrict__ C) {
  const int lane = threadIdx.x & 31;
  const int wave = threadIdx.x >> 5;
  const int kg = lane >> 4, ln = lane & 15;
  const int m0 = blockIdx.y * 32;
  const int n0 = (blockIdx.x * 4 + wave) * 64;

  const unsigned short* A0 = A + (size_t)(m0 + ln) * D_SZ;
  const unsigned short* A1 = A0 + (size_t)16 * D_SZ;
  const unsigned short* W0 = W + (size_t)(n0 + ln) * D_SZ;

  v8f acc[2][4] = {{v8f{}, v8f{}, v8f{}, v8f{}}, {v8f{}, v8f{}, v8f{}, v8f{}}};

  v16bf a0 = ld_frag_row(A0, 0, kg);
  v16bf a1 = ld_frag_row(A1, 0, kg);
  v16bf b0 = ld_frag_row(W0 + (size_t)0 * 16 * D_SZ, 0, kg);
  v16bf b1 = ld_frag_row(W0 + (size_t)1 * 16 * D_SZ, 0, kg);
  v16bf b2 = ld_frag_row(W0 + (size_t)2 * 16 * D_SZ, 0, kg);
  v16bf b3 = ld_frag_row(W0 + (size_t)3 * 16 * D_SZ, 0, kg);

  for (int k0 = 0; k0 < D_SZ; k0 += 32) {
    const int kn = (k0 + 32 < D_SZ) ? k0 + 32 : 0;
    v16bf na0 = ld_frag_row(A0, kn, kg);
    v16bf na1 = ld_frag_row(A1, kn, kg);
    v16bf nb0 = ld_frag_row(W0 + (size_t)0 * 16 * D_SZ, kn, kg);
    v16bf nb1 = ld_frag_row(W0 + (size_t)1 * 16 * D_SZ, kn, kg);
    v16bf nb2 = ld_frag_row(W0 + (size_t)2 * 16 * D_SZ, kn, kg);
    v16bf nb3 = ld_frag_row(W0 + (size_t)3 * 16 * D_SZ, kn, kg);

    acc[0][0] = wmma_bf16(a0, b0, acc[0][0]);
    acc[1][0] = wmma_bf16(a1, b0, acc[1][0]);
    acc[0][1] = wmma_bf16(a0, b1, acc[0][1]);
    acc[1][1] = wmma_bf16(a1, b1, acc[1][1]);
    acc[0][2] = wmma_bf16(a0, b2, acc[0][2]);
    acc[1][2] = wmma_bf16(a1, b2, acc[1][2]);
    acc[0][3] = wmma_bf16(a0, b3, acc[0][3]);
    acc[1][3] = wmma_bf16(a1, b3, acc[1][3]);

    a0 = na0; a1 = na1; b0 = nb0; b1 = nb1; b2 = nb2; b3 = nb3;
  }

#pragma unroll
  for (int t = 0; t < 4; ++t)
#pragma unroll
    for (int mi = 0; mi < 2; ++mi)
#pragma unroll
      for (int r = 0; r < 8; ++r)
        C[(size_t)(m0 + mi * 16 + r + kg * 8) * D_SZ + n0 + t * 16 + ln] =
            acc[mi][t][r];
}

// ---------------------------------------------------------------------------
extern "C" void kernel_launch(void* const* d_in, const int* in_sizes, int n_in,
                              void* d_out, int out_size, void* d_ws,
                              size_t ws_size, hipStream_t stream) {
  const float* x     = (const float*)d_in[0];
  const float* w_qkv = (const float*)d_in[1];
  const float* w_out = (const float*)d_in[2];
  float* out = (float*)d_out;

  const size_t nX    = (size_t)M_SZ * D_SZ;
  const size_t nWqkv = (size_t)E_SZ * D_SZ;
  const size_t nWout = (size_t)D_SZ * D_SZ;
  const size_t nQKV  = (size_t)M_SZ * D_SZ;

  unsigned short* xb    = (unsigned short*)d_ws;
  unsigned short* wqkvb = xb + nX;
  unsigned short* woutb = wqkvb + nWqkv;
  unsigned short* Qb    = woutb + nWout;
  unsigned short* Kb    = Qb + nQKV;
  unsigned short* Vtb   = Kb + nQKV;
  unsigned short* Ob    = Vtb + nQKV;

  ca_cvt_bf16<<<4096, 256, 0, stream>>>(x, xb, (int)nX);
  ca_cvt_bf16<<<4096, 256, 0, stream>>>(w_qkv, wqkvb, (int)nWqkv);
  ca_cvt_bf16<<<2048, 256, 0, stream>>>(w_out, woutb, (int)nWout);

  ca_gemm_qkv<<<dim3(E_SZ / 256, M_SZ / 32), 128, 0, stream>>>(
      xb, wqkvb, Qb, Kb, Vtb);

  ca_attn<<<dim3(T_SZ / 64, H_SZ, B_SZ), 128, 0, stream>>>(Qb, Kb, Vtb, Ob);

  ca_gemm_out<<<dim3(D_SZ / 256, M_SZ / 32), 128, 0, stream>>>(Ob, woutb, out);
}
